// QuantLinear_51110110822460
// MI455X (gfx1250) — 8-bit matrix operand order repaired and checked by reading, compiled, not run on hardware
//
#include <hip/hip_runtime.h>


typedef float v4f __attribute__((ext_vector_type(4)));
typedef int   v2i __attribute__((ext_vector_type(2)));
typedef int   v4i __attribute__((ext_vector_type(4)));
typedef int   v8i __attribute__((ext_vector_type(8)));

#define M_DIM 8192
#define K_DIM 4096
#define N_DIM 4096

#define BM 128
#define BN 128
#define BK 128
#define NKSTEPS (K_DIM / BK)
#define LDSW 36
#define ABYTES (BM * LDSW * 4)
#define BUFBYTES (2 * ABYTES)

__global__ __launch_bounds__(256)
void quant_act_kernel(const float* __restrict__ x, unsigned char* __restrict__ xq,
                      int* __restrict__ rsx, const float* __restrict__ sp,
                      const int* __restrict__ zpp) {
  const int row = blockIdx.x;
  const int t   = threadIdx.x;
  const float inv_s = 1.0f / sp[0];
  const float zp    = (float)zpp[0];

  __shared__ int ssum;
  if (t == 0) ssum = 0;
  __syncthreads();

  const float* src = x + (size_t)row * K_DIM + t * 16;
  int acc = 0;
  unsigned int packed[4];
#pragma unroll
  for (int i = 0; i < 4; ++i) {
    v4f v = __builtin_nontemporal_load((const v4f*)src + i);
    unsigned int q0 = (unsigned int)fminf(fmaxf(rintf(v.x * inv_s) + zp, 0.0f), 255.0f);
    unsigned int q1 = (unsigned int)fminf(fmaxf(rintf(v.y * inv_s) + zp, 0.0f), 255.0f);
    unsigned int q2 = (unsigned int)fminf(fmaxf(rintf(v.z * inv_s) + zp, 0.0f), 255.0f);
    unsigned int q3 = (unsigned int)fminf(fmaxf(rintf(v.w * inv_s) + zp, 0.0f), 255.0f);
    packed[i] = q0 | (q1 << 8) | (q2 << 16) | (q3 << 24);
    acc += (int)(q0 + q1 + q2 + q3);
  }
  v4i pk = { (int)packed[0], (int)packed[1], (int)packed[2], (int)packed[3] };
  *(v4i*)(xq + (size_t)row * K_DIM + t * 16) = pk;

  atomicAdd(&ssum, acc);
  __syncthreads();
  if (t == 0) rsx[row] = ssum;
}

__global__ __launch_bounds__(256)
void quant_wt_kernel(const float* __restrict__ wt, signed char* __restrict__ wq,
                     int* __restrict__ rsw, const float* __restrict__ sp,
                     const int* __restrict__ zpp) {
  const int row = blockIdx.x;
  const int t   = threadIdx.x;
  const float inv_s = 1.0f / sp[0];
  const float zp    = (float)zpp[0];

  __shared__ int ssum;
  if (t == 0) ssum = 0;
  __syncthreads();

  const float* src = wt + (size_t)row * K_DIM + t * 16;
  int acc = 0;
  unsigned int packed[4];
#pragma unroll
  for (int i = 0; i < 4; ++i) {
    v4f v = __builtin_nontemporal_load((const v4f*)src + i);
    int q0 = (int)fminf(fmaxf(rintf(v.x * inv_s) + zp, -128.0f), 127.0f);
    int q1 = (int)fminf(fmaxf(rintf(v.y * inv_s) + zp, -128.0f), 127.0f);
    int q2 = (int)fminf(fmaxf(rintf(v.z * inv_s) + zp, -128.0f), 127.0f);
    int q3 = (int)fminf(fmaxf(rintf(v.w * inv_s) + zp, -128.0f), 127.0f);
    packed[i] = (unsigned int)(q0 & 255) | ((unsigned int)(q1 & 255) << 8) |
                ((unsigned int)(q2 & 255) << 16) | ((unsigned int)(q3 & 255) << 24);
    acc += q0 + q1 + q2 + q3;
  }
  v4i pk = { (int)packed[0], (int)packed[1], (int)packed[2], (int)packed[3] };
  *(v4i*)(wq + (size_t)row * K_DIM + t * 16) = pk;

  atomicAdd(&ssum, acc);
  __syncthreads();
  if (t == 0) rsw[row] = ssum;
}

__global__ __launch_bounds__(256)
void gemm_iu8_kernel(const unsigned char* __restrict__ xq,
                     const signed char* __restrict__ wq,
                     const int* __restrict__ rsx, const int* __restrict__ rsw,
                     const float* __restrict__ bias, float* __restrict__ out,
                     const float* __restrict__ sxp, const float* __restrict__ swp,
                     const int* __restrict__ zxp, const int* __restrict__ zwp) {
  __shared__ int lds[2 * (BM + BN) * LDSW];

  const int t    = threadIdx.x;
  const int lane = t & 31;
  const int w    = t >> 5;
  const int wm   = w >> 2;
  const int wn   = w & 3;
  const int g    = lane >> 4;
  const int mr   = lane & 15;

  const int mBase = blockIdx.y * BM;
  const int nBase = blockIdx.x * BN;

  const unsigned ldsBase = (unsigned)(unsigned long long)(&lds[0]);
  unsigned           aOff[4], bOff[4];
  unsigned long long aGa[4],  bGa[4];
#pragma unroll
  for (int r = 0; r < 4; ++r) {
    const int c   = t + 256 * r;
    const int row = c >> 3;
    const int kc  = (c & 7) * 16;
    aOff[r] = (unsigned)(row * (LDSW * 4) + kc);
    bOff[r] = aOff[r] + ABYTES;
    aGa[r]  = (unsigned long long)(xq + (size_t)(mBase + row) * K_DIM + kc);
    bGa[r]  = (unsigned long long)(wq + (size_t)(nBase + row) * K_DIM + kc);
  }

  auto issue_slab = [&](int k0, int sel) {
#pragma unroll
    for (int r = 0; r < 4; ++r) {
      unsigned la = ldsBase + sel * BUFBYTES + aOff[r];
      unsigned long long ga = aGa[r] + (unsigned)k0;
      asm volatile("global_load_async_to_lds_b128 %0, %1, off"
                   :: "v"(la), "v"(ga) : "memory");
      unsigned lb = ldsBase + sel * BUFBYTES + bOff[r];
      unsigned long long gb = bGa[r] + (unsigned)k0;
      asm volatile("global_load_async_to_lds_b128 %0, %1, off"
                   :: "v"(lb), "v"(gb) : "memory");
    }
  };

  v8i acc[4][2];
#pragma unroll
  for (int i = 0; i < 4; ++i)
#pragma unroll
    for (int j = 0; j < 2; ++j)
      acc[i][j] = (v8i){0, 0, 0, 0, 0, 0, 0, 0};

  issue_slab(0, 0);

  for (int ks = 0; ks < NKSTEPS; ++ks) {
    const int sel = ks & 1;
    if (ks + 1 < NKSTEPS) {
      issue_slab((ks + 1) * BK, sel ^ 1);
      asm volatile("s_wait_asynccnt 0x8" ::: "memory");
    } else {
      asm volatile("s_wait_asynccnt 0x0" ::: "memory");
    }
    __syncthreads();

    const int* As = lds + sel * ((BM + BN) * LDSW);
    const int* Bs = As + BM * LDSW;

#pragma unroll
    for (int kk = 0; kk < 2; ++kk) {
      v8i afrag[4];
#pragma unroll
      for (int mt = 0; mt < 4; ++mt) {
        const int* rowp = &As[(wm * 64 + mt * 16 + mr) * LDSW + kk * 16 + 4 * g];
        v4i p0 = *(const v4i*)(rowp + 0);
        v4i p1 = *(const v4i*)(rowp + 8);
        afrag[mt] = (v8i){p0.x, p0.y, p0.z, p0.w, p1.x, p1.y, p1.z, p1.w};
      }
      v8i bfrag[2];
#pragma unroll
      for (int nt = 0; nt < 2; ++nt) {
        const int* rowp = &Bs[(wn * 32 + nt * 16 + mr) * LDSW + kk * 16 + 4 * g];
        v4i q0 = *(const v4i*)(rowp + 0);
        v4i q1 = *(const v4i*)(rowp + 8);
        bfrag[nt] = (v8i){q0.x, q0.y, q0.z, q0.w, q1.x, q1.y, q1.z, q1.w};
      }
#pragma unroll
      for (int mt = 0; mt < 4; ++mt)
#pragma unroll
        for (int nt = 0; nt < 2; ++nt)
          acc[mt][nt] = __builtin_amdgcn_wmma_i32_16x16x64_iu8(
               false, afrag[mt],  true, bfrag[nt],
              acc[mt][nt],  false,  false);
    }
    __syncthreads();
  }

  const float s   = sxp[0] * swp[0];
  const float zx  = (float)zxp[0];
  const float zw  = (float)zwp[0];
  const float kzz = (float)K_DIM * zx * zw;

#pragma unroll
  for (int mt = 0; mt < 4; ++mt) {
    float rx[8];
#pragma unroll
    for (int r = 0; r < 8; ++r)
      rx[r] = (float)rsx[mBase + wm * 64 + mt * 16 + 8 * g + r];
#pragma unroll
    for (int nt = 0; nt < 2; ++nt) {
      const int col  = nBase + wn * 32 + nt * 16 + mr;
      const float bv = bias[col];
      const float rw = (float)rsw[col];
#pragma unroll
      for (int r = 0; r < 8; ++r) {
        const int row = mBase + wm * 64 + mt * 16 + 8 * g + r;
        const float v = s * ((float)acc[mt][nt][r] - zw * rx[r] - zx * rw + kzz) + bv;
        __builtin_nontemporal_store(v, out + (size_t)row * N_DIM + col);
      }
    }
  }
}

extern "C" void kernel_launch(void* const* d_in, const int* in_sizes, int n_in,
                              void* d_out, int out_size, void* d_ws, size_t ws_size,
                              hipStream_t stream) {
  const float* x        = (const float*)d_in[0];
  const float* weight   = (const float*)d_in[1];
  const float* bias     = (const float*)d_in[2];
  const float* in_scale = (const float*)d_in[3];
  const float* wt_scale = (const float*)d_in[4];
  const int*   in_zp    = (const int*)d_in[5];
  const int*   wt_zp    = (const int*)d_in[6];
  float*       out      = (float*)d_out;

  unsigned char* xq = (unsigned char*)d_ws;
  signed char*   wq = (signed char*)d_ws + (size_t)M_DIM * K_DIM;
  int* rsx = (int*)((char*)d_ws + (size_t)M_DIM * K_DIM + (size_t)N_DIM * K_DIM);
  int* rsw = rsx + M_DIM;

  quant_act_kernel<<<M_DIM, 256, 0, stream>>>(x, xq, rsx, in_scale, in_zp);
  quant_wt_kernel<<<N_DIM, 256, 0, stream>>>(weight, wq, rsw, wt_scale, wt_zp);

  dim3 grid(N_DIM / BN, M_DIM / BM);
  gemm_iu8_kernel<<<grid, 256, 0, stream>>>(xq, wq, rsx, rsw, bias, out,
                                            in_scale, wt_scale, in_zp, wt_zp);
}
